// RAFTStereo_2748779069525
// MI455X (gfx1250) — matrix operand order repaired and checked by reading, not compiled, not run on hardware
//
#include <hip/hip_runtime.h>
#include <math.h>

typedef __attribute__((ext_vector_type(16))) _Float16 v16h;
typedef __attribute__((ext_vector_type(8)))  _Float16 v8h;
typedef __attribute__((ext_vector_type(8)))  float    v8f;
typedef __attribute__((ext_vector_type(4)))  unsigned int u32x4;
typedef __attribute__((ext_vector_type(8)))  int i32x8;
typedef __attribute__((ext_vector_type(4)))  int i32x4;
typedef _Float16 h16;

#define CDIV(a,b) (((a)+(b)-1)/(b))
#define BNS 0.9999950000374997f   /* 1/sqrt(1+1e-5) */

#if __has_builtin(__builtin_amdgcn_tensor_load_to_lds) && \
    __has_builtin(__builtin_amdgcn_s_wait_tensorcnt)
#define HAVE_TDM 1
#else
#define HAVE_TDM 0
#endif

// LDS strides (halves). 16B-aligned, bank-conflict-free.
#define A_STRIDE 40   /* 32 used + 8 pad halves  (80B = 20 dwords)  */
#define B_STRIDE 24   /* 16 used + 8 pad halves  (48B = 12 dwords)  */

// Global-k -> B LDS slot mapping. A lane group g reads linear halves
// [16g..16g+15] of its A row (contiguous k = 16g+s); B must use the same
// order, so kappa = linear k.
__device__ __forceinline__ int kmap(int k) {
  return k;
}

// ======================= device: WMMA micro-kernel =======================
// Per wave: 16(M) x 64(N) x 32(K) -> 4 x v_wmma_f32_16x16x32_f16.
// All fragments contiguous in LDS -> 2x ds_load_b128 each.
__device__ __forceinline__ void mm_step(const h16* Ap, const h16* Bp,
                                        int wave, int lane, v8f acc[4]) {
  const int am = lane & 15;
  const int g  = lane >> 4;
  const h16* arow = Ap + (size_t)(wave * 16 + am) * A_STRIDE + g * 16;
  v8h alo = *(const v8h*)(arow);
  v8h ahi = *(const v8h*)(arow + 8);
  v16h a = __builtin_shufflevector(alo, ahi, 0,1,2,3,4,5,6,7,8,9,10,11,12,13,14,15);
  v16h bf[4];
#pragma unroll
  for (int t = 0; t < 4; ++t) {
    const h16* bseg = Bp + (size_t)(g * 64 + t * 16 + am) * B_STRIDE;
    v8h blo = *(const v8h*)(bseg);
    v8h bhi = *(const v8h*)(bseg + 8);
    bf[t] = __builtin_shufflevector(blo, bhi, 0,1,2,3,4,5,6,7,8,9,10,11,12,13,14,15);
  }
#pragma unroll
  for (int t = 0; t < 4; ++t)
    acc[t] = __builtin_amdgcn_wmma_f32_16x16x32_f16(
        false, a, false, bf[t], (short)0, acc[t], false, false);
}

#if HAVE_TDM
// Issue one TDM DMA: linear 128x32-half weight tile, row pad 4 dwords every
// 16 dwords (-> A_STRIDE rows); OOB rows/cols zero-fill. Call from wave 0.
__device__ __forceinline__ void convA_tdm(const h16* Wt, h16* Adst,
                                          int K, int Cout, int m0, int k0) {
  unsigned lds_base = (unsigned)(unsigned long long)(size_t)(void*)Adst;
  unsigned long long gaddr = (unsigned long long)(size_t)(const void*)Wt +
      ((unsigned long long)(unsigned)m0 * (unsigned)K + (unsigned)k0) * 2ull;
  unsigned dim0 = (unsigned)(K - k0);     // remaining K (halves)
  unsigned dim1 = (unsigned)(Cout - m0);  // remaining rows
  u32x4 g0;
  g0[0] = 1u;                                   // count=1 (valid), user mode
  g0[1] = lds_base;                             // LDS byte address
  g0[2] = (unsigned)(gaddr & 0xffffffffull);    // global addr [31:0]
  g0[3] = (unsigned)((gaddr >> 32) & 0x01ffffffull) | (2u << 30); // [56:32]|type=2
  i32x8 g1;
  g1[0] = (int)((1u << 16) |   // data_size = 2 bytes
                (1u << 20) |   // pad_enable
                (3u << 22) |   // pad_interval: 16 dwords
                (3u << 25));   // pad_amount: 4 dwords
  g1[1] = (int)((dim0 & 0xffffu) << 16);                    // tensor_dim0 lo
  g1[2] = (int)((dim0 >> 16) | ((dim1 & 0xffffu) << 16));   // dim0 hi | dim1 lo
  g1[3] = (int)((dim1 >> 16) | (32u << 16));                // dim1 hi | tile_dim0=32
  g1[4] = (int)128;                                         // tile_dim1=128
  g1[5] = (int)(unsigned)K;                                 // tensor_dim0_stride lo
  g1[6] = 0;
  g1[7] = 0;
  i32x4 g2 = {0, 0, 0, 0};
  i32x4 g3 = {0, 0, 0, 0};
#if __clang_major__ >= 23
  i32x8 g4 = {0, 0, 0, 0, 0, 0, 0, 0};
  __builtin_amdgcn_tensor_load_to_lds(g0, g1, g2, g3, g4, 0);
#else
  __builtin_amdgcn_tensor_load_to_lds(g0, g1, g2, g3, 0);
#endif
}
#else
__device__ __forceinline__ void convA_manual(const h16* __restrict__ Wt, h16* Adst,
                                             int K, int Cout, int m0, int k0, int tid) {
  for (int i = tid; i < 128 * 32; i += 256) {
    int m = i >> 5, k = i & 31;
    int gm = m0 + m, gk = k0 + k;
    h16 v = (h16)0.0f;
    if (gm < Cout && gk < K) {
      v = Wt[(size_t)gm * K + gk];
      if (k == 0 && k0 + 64 <= K)
        __builtin_prefetch(&Wt[(size_t)gm * K + k0 + 64], 0, 0);
    }
    Adst[m * A_STRIDE + k] = v;
  }
}
#endif

// im2col gather split into load-phase (values + LDS slots into registers) and
// store-phase so the global-load latency overlaps the WMMAs in between.
__device__ __forceinline__ void convB_load(const h16* __restrict__ X,
    int n, int Cin, int H, int Wd, int Ho, int Wo, int ksz, int kk,
    int stride, int pad, int K, int k0, int p0, int HWo, int tid,
    h16 val[8], int slot[8]) {
#pragma unroll
  for (int u = 0; u < 8; ++u) {
    int i = tid + u * 256;
    int k = i >> 6, nn = i & 63;
    int gk = k0 + k, pix = p0 + nn;
    h16 v = (h16)0.0f;
    if (gk < K && pix < HWo) {
      int ci = gk / kk, r = gk - ci * kk;
      int kh = r / ksz, kw = r - kh * ksz;
      int oy = pix / Wo, ox = pix - oy * Wo;
      int iy = oy * stride - pad + kh;
      int ix = ox * stride - pad + kw;
      if (iy >= 0 && iy < H && ix >= 0 && ix < Wd)
        v = X[(((size_t)n * Cin + ci) * H + iy) * Wd + ix];
    }
    int kp = kmap(k);
    val[u]  = v;
    slot[u] = ((kp >> 4) * 64 + nn) * B_STRIDE + (kp & 15);
  }
}
__device__ __forceinline__ void convB_store(h16* Bdst, const h16 val[8],
                                            const int slot[8]) {
#pragma unroll
  for (int u = 0; u < 8; ++u) Bdst[slot[u]] = val[u];
}

// ======================= implicit-GEMM convolution =======================
// X: (N,Cin,H,W) f16. Wt: (Cout, Cin*k*k) f16 row-major. Y: (N,Cout,Ho,Wo) f16.
// Block: 256 threads (8 waves), tile 128(M=Cout) x 64(N=pixels), K-step 32.
// Double-buffered pipeline: TDM DMA + global gathers for step k+1 run under
// the WMMAs of step k; one barrier per K-step.
// y = [relu]( alpha * (conv + bias) )
__global__ __launch_bounds__(256) void k_conv_igemm(
    const h16* __restrict__ X, const h16* __restrict__ Wt,
    const float* __restrict__ bias, h16* __restrict__ Y,
    int N, int Cin, int H, int Wd, int Cout, int Ho, int Wo,
    int ksz, int stride, int pad, float alpha, int relu) {
  __shared__ h16 Ap[2][128 * A_STRIDE];
  __shared__ h16 Bp[2][128 * B_STRIDE];
  const int tid = threadIdx.x, wave = tid >> 5, lane = tid & 31;
  const int n  = blockIdx.z;
  const int m0 = blockIdx.y * 128;
  const int p0 = blockIdx.x * 64;
  const int K  = Cin * ksz * ksz;
  const int HWo = Ho * Wo;
  const int kk = ksz * ksz;
  v8f acc[4] = {};
  int cur = 0;

  // ---- prologue: stage K-step 0 ----
#if HAVE_TDM
  if (wave == 0) convA_tdm(Wt, Ap[0], K, Cout, m0, 0);
#else
  convA_manual(Wt, Ap[0], K, Cout, m0, 0, tid);
#endif
  {
    h16 bv[8]; int bs[8];
    convB_load(X, n, Cin, H, Wd, Ho, Wo, ksz, kk, stride, pad, K, 0, p0, HWo,
               tid, bv, bs);
    convB_store(Bp[0], bv, bs);
  }
#if HAVE_TDM
  if (wave == 0) __builtin_amdgcn_s_wait_tensorcnt((short)0);
#endif
  __syncthreads();

  for (int k0 = 0; k0 < K; k0 += 32) {
    int nk = k0 + 32;
    h16 bv[8]; int bs[8];
    if (nk < K) {
#if HAVE_TDM
      if (wave == 0) convA_tdm(Wt, Ap[cur ^ 1], K, Cout, m0, nk);  // async DMA
#else
      convA_manual(Wt, Ap[cur ^ 1], K, Cout, m0, nk, tid);
#endif
      convB_load(X, n, Cin, H, Wd, Ho, Wo, ksz, kk, stride, pad, K, nk, p0,
                 HWo, tid, bv, bs);                                // loads in flight
    }
    mm_step(Ap[cur], Bp[cur], wave, lane, acc);                    // overlaps loads
    if (nk < K) {
      convB_store(Bp[cur ^ 1], bv, bs);
#if HAVE_TDM
      if (wave == 0) __builtin_amdgcn_s_wait_tensorcnt((short)0);
#endif
      __syncthreads();
    }
    cur ^= 1;
  }
  const int mrow = wave << 4;
#pragma unroll
  for (int t = 0; t < 4; ++t) {
    int pix = p0 + t * 16 + (lane & 15);
    if (pix >= HWo) continue;
#pragma unroll
    for (int r = 0; r < 8; ++r) {
      int gm = m0 + mrow + r + ((lane >> 4) << 3);
      if (gm < Cout) {
        float v = alpha * (acc[t][r] + bias[gm]);
        if (relu) v = fmaxf(v, 0.0f);
        Y[((size_t)n * Cout + gm) * HWo + pix] = (h16)v;
      }
    }
  }
}

// ======================= correlation volume GEMM =======================
// corr[h][i][j] = scale * sum_d f1[d,h,i] * f2[d,h,j]; tile 128(i) x 64(j),
// double-buffered (one barrier per K-step).
__global__ __launch_bounds__(256) void k_corr_wmma(
    const h16* __restrict__ f1, const h16* __restrict__ f2,
    float* __restrict__ corr, int C, int Hh, int Wd, float scale) {
  __shared__ h16 Ap[2][128 * A_STRIDE];
  __shared__ h16 Bp[2][128 * B_STRIDE];
  const int tid = threadIdx.x, wave = tid >> 5, lane = tid & 31;
  const int h  = blockIdx.z;
  const int i0 = blockIdx.y * 128, j0 = blockIdx.x * 64;
  const int HW = Hh * Wd;
  v8f acc[4] = {};
  int cur = 0;

  for (int t = tid; t < 128 * 32; t += 256) {
    int m = t >> 5, k = t & 31;
    Ap[0][m * A_STRIDE + k] = f1[(size_t)k * HW + h * Wd + i0 + m];
  }
  for (int t = tid; t < 32 * 64; t += 256) {
    int k = t >> 6, nn = t & 63;
    int kp = kmap(k);
    Bp[0][((kp >> 4) * 64 + nn) * B_STRIDE + (kp & 15)] =
        f2[(size_t)k * HW + h * Wd + j0 + nn];
  }
  __syncthreads();

  for (int k0 = 0; k0 < C; k0 += 32) {
    int nk = k0 + 32;
    if (nk < C) {
      for (int t = tid; t < 128 * 32; t += 256) {
        int m = t >> 5, k = t & 31;
        Ap[cur ^ 1][m * A_STRIDE + k] = f1[(size_t)(nk + k) * HW + h * Wd + i0 + m];
      }
      for (int t = tid; t < 32 * 64; t += 256) {
        int k = t >> 6, nn = t & 63;
        int kp = kmap(k);
        Bp[cur ^ 1][((kp >> 4) * 64 + nn) * B_STRIDE + (kp & 15)] =
            f2[(size_t)(nk + k) * HW + h * Wd + j0 + nn];
      }
    }
    mm_step(Ap[cur], Bp[cur], wave, lane, acc);
    if (nk < C) __syncthreads();
    cur ^= 1;
  }
  const int mrow = wave << 4;
#pragma unroll
  for (int t = 0; t < 4; ++t) {
    int j = j0 + t * 16 + (lane & 15);
#pragma unroll
    for (int r = 0; r < 8; ++r) {
      int i = i0 + mrow + r + ((lane >> 4) << 3);
      corr[((size_t)h * Wd + i) * Wd + j] = scale * acc[t][r];
    }
  }
}

// ======================= elementwise / support kernels =======================
__global__ void k_pack(const float* __restrict__ w, h16* __restrict__ o, int n) {
  int i = blockIdx.x * blockDim.x + threadIdx.x;
  if (i < n) o[i] = (h16)w[i];
}
__global__ void k_img_norm(const float* __restrict__ x, h16* __restrict__ y, int n) {
  int i = blockIdx.x * blockDim.x + threadIdx.x;
  if (i < n) y[i] = (h16)(2.0f * (x[i] / 255.0f) - 1.0f);
}
__global__ void k_copy16(const h16* __restrict__ x, h16* __restrict__ y, int n) {
  int i = blockIdx.x * blockDim.x + threadIdx.x;
  if (i < n) y[i] = x[i];
}
__global__ void k_tanh16(const h16* __restrict__ x, h16* __restrict__ y, int n) {
  int i = blockIdx.x * blockDim.x + threadIdx.x;
  if (i < n) y[i] = (h16)tanhf((float)x[i]);
}
__global__ void k_add_relu(const h16* __restrict__ a, h16* __restrict__ y, int n) {
  int i = blockIdx.x * blockDim.x + threadIdx.x;
  if (i < n) { float v = (float)a[i] + (float)y[i]; y[i] = (h16)fmaxf(v, 0.0f); }
}
__global__ __launch_bounds__(256) void k_inorm(h16* __restrict__ x, int HW, int relu) {
  h16* p = x + (size_t)blockIdx.x * HW;
  __shared__ float ss[256], sq[256];
  float s = 0.f, q = 0.f;
  for (int i = threadIdx.x; i < HW; i += 256) { float v = (float)p[i]; s += v; q += v*v; }
  ss[threadIdx.x] = s; sq[threadIdx.x] = q; __syncthreads();
  for (int o = 128; o > 0; o >>= 1) {
    if (threadIdx.x < o) { ss[threadIdx.x] += ss[threadIdx.x+o]; sq[threadIdx.x] += sq[threadIdx.x+o]; }
    __syncthreads();
  }
  float m  = ss[0] / (float)HW;
  float vv = sq[0] / (float)HW - m * m;
  float inv = rsqrtf(vv + 1e-5f);
  for (int i = threadIdx.x; i < HW; i += 256) {
    float v = ((float)p[i] - m) * inv;
    if (relu) v = fmaxf(v, 0.0f);
    p[i] = (h16)v;
  }
}
__global__ void k_pool2x(const h16* __restrict__ x, h16* __restrict__ y,
                         int C, int H, int W, int Ho, int Wo) {
  int i = blockIdx.x * blockDim.x + threadIdx.x;
  int tot = C * Ho * Wo;
  if (i >= tot) return;
  int c = i / (Ho*Wo), r = i - c*Ho*Wo;
  int oy = r / Wo, ox = r - oy*Wo;
  float s = 0.f;
  for (int dy = 0; dy < 3; dy++)
    for (int dx = 0; dx < 3; dx++) {
      int iy = 2*oy - 1 + dy, ix = 2*ox - 1 + dx;
      if (iy >= 0 && iy < H && ix >= 0 && ix < W)
        s += (float)x[((size_t)c*H + iy)*W + ix];
    }
  y[i] = (h16)(s / 9.0f);
}
__global__ void k_interp(const h16* __restrict__ x, h16* __restrict__ y,
                         int C, int Hi, int Wi, int Ho, int Wo) {
  int i = blockIdx.x * blockDim.x + threadIdx.x;
  int tot = C * Ho * Wo;
  if (i >= tot) return;
  int c = i / (Ho*Wo), r = i - c*Ho*Wo;
  int oy = r / Wo, ox = r - oy*Wo;
  float fy = (float)oy * (float)(Hi - 1) / (float)(Ho - 1);
  float fx = (float)ox * (float)(Wi - 1) / (float)(Wo - 1);
  int y0 = (int)floorf(fy); if (y0 > Hi-2) y0 = Hi-2; if (y0 < 0) y0 = 0;
  int x0 = (int)floorf(fx); if (x0 > Wi-2) x0 = Wi-2; if (x0 < 0) x0 = 0;
  float wy = fy - y0, wx = fx - x0;
  const h16* b = x + (size_t)c*Hi*Wi;
  float v00 = (float)b[(size_t)y0*Wi + x0],     v01 = (float)b[(size_t)y0*Wi + x0 + 1];
  float v10 = (float)b[(size_t)(y0+1)*Wi + x0], v11 = (float)b[(size_t)(y0+1)*Wi + x0 + 1];
  float v = (v00*(1-wx)+v01*wx)*(1-wy) + (v10*(1-wx)+v11*wx)*wy;
  y[i] = (h16)v;
}
__global__ void k_pyr_down(const float* __restrict__ in, float* __restrict__ out,
                           int rows, int W2o) {
  int i = blockIdx.x * blockDim.x + threadIdx.x;
  if (i >= rows * W2o) return;
  int r = i / W2o, j = i - r * W2o;
  out[i] = 0.5f * (in[(size_t)r * (2*W2o) + 2*j] + in[(size_t)r * (2*W2o) + 2*j + 1]);
}
__global__ void k_lookup(const float* __restrict__ L0, const float* __restrict__ L1,
                         const float* __restrict__ L2, const float* __restrict__ L3,
                         const float* __restrict__ cx, h16* __restrict__ out,
                         int Hh, int Wd) {
  int idx = blockIdx.x * blockDim.x + threadIdx.x;
  int HW = Hh * Wd;
  if (idx >= HW) return;
  float c = cx[idx];
  const float* pyr[4] = {L0, L1, L2, L3};
  int W2 = Wd;
  for (int l = 0; l < 4; l++) {
    const float* P = pyr[l] + (size_t)idx * W2;
    float xl = c / (float)(1 << l);
    for (int d = 0; d < 9; d++) {
      float xq = xl + (float)(d - 4);
      float x0f = floorf(xq);
      float w = xq - x0f;
      int x0 = (int)x0f, x1 = x0 + 1;
      float v0 = (x0 >= 0 && x0 < W2) ? P[x0] : 0.0f;
      float v1 = (x1 >= 0 && x1 < W2) ? P[x1] : 0.0f;
      out[(size_t)(l*9 + d) * HW + idx] = (h16)(v0*(1.0f - w) + v1*w);
    }
    W2 >>= 1;
  }
}
__device__ __forceinline__ float sigm(float v) { return 1.0f / (1.0f + expf(-v)); }
__global__ void k_gru_zr(const h16* az, const h16* ar, const h16* cz, const h16* cr,
                         const h16* h, h16* z, h16* rh, int n) {
  int i = blockIdx.x * blockDim.x + threadIdx.x;
  if (i >= n) return;
  float zz = sigm((float)az[i] + (float)cz[i]);
  float rr = sigm((float)ar[i] + (float)cr[i]);
  z[i]  = (h16)zz;
  rh[i] = (h16)(rr * (float)h[i]);
}
__global__ void k_gru_out(const h16* h, const h16* z, const h16* aq, const h16* cq,
                          h16* hn, int n) {
  int i = blockIdx.x * blockDim.x + threadIdx.x;
  if (i >= n) return;
  float q = tanhf((float)aq[i] + (float)cq[i]);
  float zz = (float)z[i];
  hn[i] = (h16)((1.0f - zz) * (float)h[i] + zz * q);
}
__global__ void k_init_cx(float* cx, int W4, int n) {
  int i = blockIdx.x * blockDim.x + threadIdx.x;
  if (i < n) cx[i] = (float)(i % W4);
}
__global__ void k_update_cx(float* cx, const h16* delta, int n) {
  int i = blockIdx.x * blockDim.x + threadIdx.x;
  if (i < n) cx[i] += (float)delta[i];
}
__global__ void k_make_flow16(const float* cx, h16* fl, int W4, int HW) {
  int i = blockIdx.x * blockDim.x + threadIdx.x;
  if (i >= HW) return;
  fl[i] = (h16)(cx[i] - (float)(i % W4));
  fl[HW + i] = (h16)0.0f;
}
__global__ void k_upsample(const float* __restrict__ cx, const h16* __restrict__ mask,
                           float* __restrict__ out, int H4, int W4) {
  int idx = blockIdx.x * blockDim.x + threadIdx.x;
  int WO = W4 * 4, HO = H4 * 4;
  if (idx >= HO * WO) return;
  int y = idx / WO, x = idx - y * WO;
  int h = y >> 2, fy = y & 3, w = x >> 2, fx = x & 3;
  int HW = H4 * W4;
  float lg[9], mx = -1e30f;
  for (int k = 0; k < 9; k++) {
    lg[k] = (float)mask[(size_t)(k*16 + fy*4 + fx) * HW + h * W4 + w];
    mx = fmaxf(mx, lg[k]);
  }
  float se = 0.f;
  for (int k = 0; k < 9; k++) { lg[k] = expf(lg[k] - mx); se += lg[k]; }
  float acc = 0.f;
  for (int k = 0; k < 9; k++) {
    int di = k / 3, dj = k - 3*di;
    int hh = h + di - 1, ww = w + dj - 1;
    float fv = 0.f;
    if (hh >= 0 && hh < H4 && ww >= 0 && ww < W4)
      fv = 4.0f * (cx[hh * W4 + ww] - (float)ww);
    acc += lg[k] * fv;
  }
  out[idx] = acc / se;
}

// ======================= host-side orchestration =======================
struct Conv { const float* b; const float* w; int cin, cout, k; };
struct RB   { Conv c1, c2, d; int down; };
struct GRU  { Conv q, r, z; };

static Conv gc(void* const* in, int& i, int cin, int cout, int k) {
  Conv c; c.b = (const float*)in[i++]; c.w = (const float*)in[i++];
  c.cin = cin; c.cout = cout; c.k = k; return c;
}
static RB grb(void* const* in, int& i, int cin, int cout, int down) {
  RB r; r.c1 = gc(in, i, cin, cout, 3); r.c2 = gc(in, i, cout, cout, 3);
  r.down = down; if (down) r.d = gc(in, i, cin, cout, 1); else r.d = r.c1;
  return r;
}

static void conv(hipStream_t s, h16* wp, const Conv& p, const h16* x, h16* y,
                 int N, int H, int W, int stride, float alpha, int relu,
                 int& Ho, int& Wo) {
  int K = p.cin * p.k * p.k;
  int nw = p.cout * K;
  k_pack<<<CDIV(nw, 256), 256, 0, s>>>(p.w, wp, nw);
  int pad = p.k / 2;
  Ho = (H + 2*pad - p.k) / stride + 1;
  Wo = (W + 2*pad - p.k) / stride + 1;
  dim3 g(CDIV(Ho * Wo, 64), CDIV(p.cout, 128), N);
  k_conv_igemm<<<g, 256, 0, s>>>(x, wp, p.b, y, N, p.cin, H, W, p.cout,
                                 Ho, Wo, p.k, stride, pad, alpha, relu);
}

static void resblock(hipStream_t s, h16* wp, const RB& r, int inorm,
                     const h16* x, h16* A, h16* B, int N, int& H, int& W, int stride) {
  int Ho, Wo, H2, W2;
  float al = inorm ? 1.0f : BNS;
  int fr = inorm ? 0 : 1;
  conv(s, wp, r.c1, x, A, N, H, W, stride, al, fr, Ho, Wo);
  if (inorm) k_inorm<<<N * r.c1.cout, 256, 0, s>>>(A, Ho * Wo, 1);
  conv(s, wp, r.c2, A, B, N, Ho, Wo, 1, al, fr, H2, W2);
  if (inorm) k_inorm<<<N * r.c2.cout, 256, 0, s>>>(B, H2 * W2, 1);
  const h16* sk = x;
  if (r.down) {
    conv(s, wp, r.d, x, A, N, H, W, stride, al, 0, Ho, Wo);
    if (inorm) k_inorm<<<N * r.d.cout, 256, 0, s>>>(A, Ho * Wo, 0);
    sk = A;
  }
  int n = N * r.c2.cout * H2 * W2;
  k_add_relu<<<CDIV(n, 256), 256, 0, s>>>(sk, B, n);
  H = H2; W = W2;
}

struct Scr { h16 *hx, *az, *ar, *aq, *z, *rh; };

static void gru_step(hipStream_t s, h16* wp, const GRU& g, h16* h, const h16* xin,
                     int Cx, const h16* inp, int H, int W, h16* hnew, Scr& S) {
  int HW = H * W, d1, d2;
  k_copy16<<<CDIV(128*HW, 256), 256, 0, s>>>(h, S.hx, 128*HW);
  k_copy16<<<CDIV(Cx*HW, 256), 256, 0, s>>>(xin, S.hx + (size_t)128*HW, Cx*HW);
  conv(s, wp, g.z, S.hx, S.az, 1, H, W, 1, 1.0f, 0, d1, d2);
  conv(s, wp, g.r, S.hx, S.ar, 1, H, W, 1, 1.0f, 0, d1, d2);
  const h16* cz = inp;
  const h16* cr = inp + (size_t)128*HW;
  const h16* cq = inp + (size_t)256*HW;
  k_gru_zr<<<CDIV(128*HW, 256), 256, 0, s>>>(S.az, S.ar, cz, cr, h, S.z, S.rh, 128*HW);
  k_copy16<<<CDIV(128*HW, 256), 256, 0, s>>>(S.rh, S.hx, 128*HW);
  conv(s, wp, g.q, S.hx, S.aq, 1, H, W, 1, 1.0f, 0, d1, d2);
  k_gru_out<<<CDIV(128*HW, 256), 256, 0, s>>>(h, S.z, S.aq, cq, hnew, 128*HW);
}

struct Arena {
  char* p; size_t used;
  void* take(size_t bytes) { bytes = (bytes + 255) & ~(size_t)255; void* r = p + used; used += bytes; return r; }
};

extern "C" void kernel_launch(void* const* d_in, const int* in_sizes, int n_in,
                              void* d_out, int out_size, void* d_ws, size_t ws_size,
                              hipStream_t stream) {
  (void)in_sizes; (void)n_in; (void)out_size;
  hipStream_t s = stream;

  // ---- parse params (jax pytree order: sorted dict keys, lists in order) ----
  int ix = 0;
  const float* img1 = (const float*)d_in[ix++];
  const float* img2 = (const float*)d_in[ix++];
  // cnet
  Conv cn1 = gc(d_in, ix, 3, 64, 7);
  RB cl1[2] = { grb(d_in, ix, 64, 64, 0),  grb(d_in, ix, 64, 64, 0) };
  RB cl2[2] = { grb(d_in, ix, 64, 96, 1),  grb(d_in, ix, 96, 96, 0) };
  RB cl3[2] = { grb(d_in, ix, 96, 128, 1), grb(d_in, ix, 128, 128, 0) };
  RB cl4[2] = { grb(d_in, ix, 128, 128, 1), grb(d_in, ix, 128, 128, 0) };
  RB cl5[2] = { grb(d_in, ix, 128, 128, 1), grb(d_in, ix, 128, 128, 0) };
  Conv o08c[2]; RB o08r[2];
  for (int j = 0; j < 2; j++) { o08c[j] = gc(d_in, ix, 128, 128, 3); o08r[j] = grb(d_in, ix, 128, 128, 0); }
  Conv o16c[2] = { gc(d_in, ix, 128, 128, 3), gc(d_in, ix, 128, 128, 3) };
  Conv o32c[2] = { gc(d_in, ix, 128, 128, 3), gc(d_in, ix, 128, 128, 3) };
  // fnet
  Conv fn1 = gc(d_in, ix, 3, 64, 7);
  Conv fn2 = gc(d_in, ix, 128, 256, 1);
  RB fl1[2] = { grb(d_in, ix, 64, 64, 0),  grb(d_in, ix, 64, 64, 0) };
  RB fl2[2] = { grb(d_in, ix, 64, 96, 1),  grb(d_in, ix, 96, 96, 0) };
  RB fl3[2] = { grb(d_in, ix, 96, 128, 1), grb(d_in, ix, 128, 128, 0) };
  // upd
  Conv e_conv  = gc(d_in, ix, 128, 126, 3);
  Conv e_c1    = gc(d_in, ix, 36, 64, 1);
  Conv e_c2    = gc(d_in, ix, 64, 64, 3);
  Conv e_f1    = gc(d_in, ix, 2, 64, 7);
  Conv e_f2    = gc(d_in, ix, 64, 64, 3);
  Conv fh1     = gc(d_in, ix, 128, 256, 3);
  Conv fh2     = gc(d_in, ix, 256, 2, 3);
  GRU g08 = { gc(d_in, ix, 384, 128, 3), gc(d_in, ix, 384, 128, 3), gc(d_in, ix, 384, 128, 3) };
  GRU g16 = { gc(d_in, ix, 384, 128, 3), gc(d_in, ix, 384, 128, 3), gc(d_in, ix, 384, 128, 3) };
  GRU g32 = { gc(d_in, ix, 256, 128, 3), gc(d_in, ix, 256, 128, 3), gc(d_in, ix, 256, 128, 3) };
  Conv mask1 = gc(d_in, ix, 128, 256, 3);
  Conv mask2 = gc(d_in, ix, 256, 144, 1);
  Conv zqr[3] = { gc(d_in, ix, 128, 384, 3), gc(d_in, ix, 128, 384, 3), gc(d_in, ix, 128, 384, 3) };
  const int ITERS = 3;

  // ---- workspace layout ----
  const int HWi = 384 * 512;
  const size_t HW0 = 96 * 128, HW1 = 48 * 64, HW2 = 24 * 32;
  Arena A; A.p = (char*)d_ws; A.used = 0;
  h16* fin  = (h16*)A.take((size_t)2 * 3 * HWi * 2);
  h16* S0   = (h16*)A.take((size_t)2 * 64 * HWi * 2);
  h16* S1   = (h16*)A.take((size_t)2 * 64 * HWi * 2);
  h16* S2   = (h16*)A.take((size_t)2 * 64 * HWi * 2);
  h16* fmap = (h16*)A.take((size_t)2 * 256 * HW0 * 2);
  float* pyr = (float*)A.take(HW0 * 240 * 4);
  h16* netb[3][2];
  size_t hwS[3] = { HW0, HW1, HW2 };
  for (int i = 0; i < 3; i++)
    for (int j = 0; j < 2; j++)
      netb[i][j] = (h16*)A.take((size_t)128 * hwS[i] * 2);
  h16* inpb[3];
  for (int i = 0; i < 3; i++) inpb[i] = (h16*)A.take((size_t)384 * hwS[i] * 2);
  h16* corr36 = (h16*)A.take((size_t)36 * HW0 * 2);
  float* cx   = (float*)A.take(HW0 * 4);
  h16* wp     = (h16*)A.take((size_t)600000 * 2);
  Scr SC;
  SC.hx = (h16*)A.take((size_t)384 * HW0 * 2);
  SC.az = (h16*)A.take((size_t)128 * HW0 * 2);
  SC.ar = (h16*)A.take((size_t)128 * HW0 * 2);
  SC.aq = (h16*)A.take((size_t)128 * HW0 * 2);
  SC.z  = (h16*)A.take((size_t)128 * HW0 * 2);
  SC.rh = (h16*)A.take((size_t)128 * HW0 * 2);
  h16* gx    = (h16*)A.take((size_t)256 * HW0 * 2);
  h16* ec1   = (h16*)A.take((size_t)64 * HW0 * 2);
  h16* ec2   = (h16*)A.take((size_t)64 * HW0 * 2);
  h16* ec3   = (h16*)A.take((size_t)64 * HW0 * 2);
  h16* ecat  = (h16*)A.take((size_t)128 * HW0 * 2);
  h16* mfo   = (h16*)A.take((size_t)126 * HW0 * 2);
  h16* mf    = (h16*)A.take((size_t)128 * HW0 * 2);
  h16* fl16  = (h16*)A.take((size_t)2 * HW0 * 2);
  h16* dbuf  = (h16*)A.take((size_t)256 * HW0 * 2);
  h16* delta = (h16*)A.take((size_t)2 * HW0 * 2);
  h16* maskb = (h16*)A.take((size_t)144 * HW0 * 2);
  if (A.used > ws_size) return;   // not enough scratch; bail deterministically

  float* L0 = pyr;
  float* L1 = L0 + HW0 * 128;
  float* L2 = L1 + HW0 * 64;
  float* L3 = L2 + HW0 * 32;

  int H, W, Ho, Wo, d1, d2;

  // ---- normalize images; fin holds batch-2 stack ----
  k_img_norm<<<CDIV(3*HWi, 256), 256, 0, s>>>(img1, fin, 3*HWi);
  k_img_norm<<<CDIV(3*HWi, 256), 256, 0, s>>>(img2, fin + (size_t)3*HWi, 3*HWi);

  // ---- fnet (N=2) ----
  H = 384; W = 512;
  conv(s, wp, fn1, fin, S0, 2, H, W, 1, 1.0f, 0, Ho, Wo);
  k_inorm<<<2*64, 256, 0, s>>>(S0, Ho*Wo, 1);
  H = Ho; W = Wo;
  resblock(s, wp, fl1[0], 1, S0, S1, S2, 2, H, W, 1);
  resblock(s, wp, fl1[1], 1, S2, S1, S0, 2, H, W, 1);
  resblock(s, wp, fl2[0], 1, S0, S1, S2, 2, H, W, 2);
  resblock(s, wp, fl2[1], 1, S2, S1, S0, 2, H, W, 1);
  resblock(s, wp, fl3[0], 1, S0, S1, S2, 2, H, W, 2);
  resblock(s, wp, fl3[1], 1, S2, S1, S0, 2, H, W, 1);
  conv(s, wp, fn2, S0, fmap, 2, H, W, 1, 1.0f, 0, Ho, Wo);   // 2 x 256 x 96 x 128
  const h16* fm1 = fmap;
  const h16* fm2 = fmap + (size_t)256 * HW0;

  // ---- correlation pyramid ----
  {
    dim3 g(2, 1, 96);
    k_corr_wmma<<<g, 256, 0, s>>>(fm1, fm2, L0, 256, 96, 128, 1.0f/16.0f);
    k_pyr_down<<<CDIV((int)(HW0*64), 256), 256, 0, s>>>(L0, L1, (int)HW0, 64);
    k_pyr_down<<<CDIV((int)(HW0*32), 256), 256, 0, s>>>(L1, L2, (int)HW0, 32);
    k_pyr_down<<<CDIV((int)(HW0*16), 256), 256, 0, s>>>(L2, L3, (int)HW0, 16);
  }

  // ---- cnet (N=1, bnorm fused into conv epilogue) ----
  H = 384; W = 512;
  conv(s, wp, cn1, fin, S0, 1, H, W, 1, BNS, 1, Ho, Wo); H = Ho; W = Wo;
  resblock(s, wp, cl1[0], 0, S0, S1, S2, 1, H, W, 1);
  resblock(s, wp, cl1[1], 0, S2, S1, S0, 1, H, W, 1);
  resblock(s, wp, cl2[0], 0, S0, S1, S2, 1, H, W, 2);
  resblock(s, wp, cl2[1], 0, S2, S1, S0, 1, H, W, 1);
  resblock(s, wp, cl3[0], 0, S0, S1, S2, 1, H, W, 2);
  resblock(s, wp, cl3[1], 0, S2, S1, S0, 1, H, W, 1);   // x08 in S0 @96x128
  int net_cur[3] = {0, 0, 0};
  // scale-0 heads
  {
    int Hh = H, Wh = W;
    resblock(s, wp, o08r[0], 0, S0, S1, S2, 1, Hh, Wh, 1);
    conv(s, wp, o08c[0], S2, S1, 1, Hh, Wh, 1, 1.0f, 0, d1, d2);
    k_tanh16<<<CDIV((int)(128*HW0), 256), 256, 0, s>>>(S1, netb[0][0], (int)(128*HW0));
    Hh = H; Wh = W;
    resblock(s, wp, o08r[1], 0, S0, S1, S2, 1, Hh, Wh, 1);
    conv(s, wp, o08c[1], S2, S1, 1, Hh, Wh, 1, 1.0f, 1, d1, d2);   // relu(o1)
    conv(s, wp, zqr[0], S1, inpb[0], 1, Hh, Wh, 1, 1.0f, 0, d1, d2);
  }
  resblock(s, wp, cl4[0], 0, S0, S1, S2, 1, H, W, 2);
  resblock(s, wp, cl4[1], 0, S2, S1, S0, 1, H, W, 1);   // x16 in S0 @48x64
  {
    conv(s, wp, o16c[0], S0, S1, 1, H, W, 1, 1.0f, 0, d1, d2);
    k_tanh16<<<CDIV((int)(128*HW1), 256), 256, 0, s>>>(S1, netb[1][0], (int)(128*HW1));
    conv(s, wp, o16c[1], S0, S1, 1, H, W, 1, 1.0f, 1, d1, d2);
    conv(s, wp, zqr[1], S1, inpb[1], 1, H, W, 1, 1.0f, 0, d1, d2);
  }
  resblock(s, wp, cl5[0], 0, S0, S1, S2, 1, H, W, 2);
  resblock(s, wp, cl5[1], 0, S2, S1, S0, 1, H, W, 1);   // x32 in S0 @24x32
  {
    conv(s, wp, o32c[0], S0, S1, 1, H, W, 1, 1.0f, 0, d1, d2);
    k_tanh16<<<CDIV((int)(128*HW2), 256), 256, 0, s>>>(S1, netb[2][0], (int)(128*HW2));
    conv(s, wp, o32c[1], S0, S1, 1, H, W, 1, 1.0f, 1, d1, d2);
    conv(s, wp, zqr[2], S1, inpb[2], 1, H, W, 1, 1.0f, 0, d1, d2);
  }

  // ---- iterative refinement ----
  k_init_cx<<<CDIV((int)HW0, 256), 256, 0, s>>>(cx, 128, (int)HW0);
  for (int it = 0; it < ITERS; it++) {
    h16* n0 = netb[0][net_cur[0]]; h16* n0n = netb[0][1 - net_cur[0]];
    h16* n1 = netb[1][net_cur[1]]; h16* n1n = netb[1][1 - net_cur[1]];
    h16* n2 = netb[2][net_cur[2]]; h16* n2n = netb[2][1 - net_cur[2]];

    // correlation lookup at current coords
    k_lookup<<<CDIV((int)HW0, 256), 256, 0, s>>>(L0, L1, L2, L3, cx, corr36, 96, 128);

    // n2 = gru32(net2, pool2x(net1))
    k_pool2x<<<CDIV((int)(128*HW2), 256), 256, 0, s>>>(n1, gx, 128, 48, 64, 24, 32);
    gru_step(s, wp, g32, n2, gx, 128, inpb[2], 24, 32, n2n, SC);
    net_cur[2] = 1 - net_cur[2]; n2 = n2n;

    // n1 = gru16(net1, cat(pool2x(net0), interp(n2)))
    k_pool2x<<<CDIV((int)(128*HW1), 256), 256, 0, s>>>(n0, gx, 128, 96, 128, 48, 64);
    k_interp<<<CDIV((int)(128*HW1), 256), 256, 0, s>>>(n2, gx + (size_t)128*HW1, 128, 24, 32, 48, 64);
    gru_step(s, wp, g16, n1, gx, 256, inpb[1], 48, 64, n1n, SC);
    net_cur[1] = 1 - net_cur[1]; n1 = n1n;

    // motion encoder
    conv(s, wp, e_c1, corr36, ec1, 1, 96, 128, 1, 1.0f, 1, d1, d2);
    conv(s, wp, e_c2, ec1,    ec2, 1, 96, 128, 1, 1.0f, 1, d1, d2);
    k_make_flow16<<<CDIV((int)HW0, 256), 256, 0, s>>>(cx, fl16, 128, (int)HW0);
    conv(s, wp, e_f1, fl16, ec1, 1, 96, 128, 1, 1.0f, 1, d1, d2);
    conv(s, wp, e_f2, ec1,  ec3, 1, 96, 128, 1, 1.0f, 1, d1, d2);
    k_copy16<<<CDIV((int)(64*HW0), 256), 256, 0, s>>>(ec2, ecat, (int)(64*HW0));
    k_copy16<<<CDIV((int)(64*HW0), 256), 256, 0, s>>>(ec3, ecat + (size_t)64*HW0, (int)(64*HW0));
    conv(s, wp, e_conv, ecat, mfo, 1, 96, 128, 1, 1.0f, 1, d1, d2);
    k_copy16<<<CDIV((int)(126*HW0), 256), 256, 0, s>>>(mfo, mf, (int)(126*HW0));
    k_copy16<<<CDIV((int)(2*HW0), 256), 256, 0, s>>>(fl16, mf + (size_t)126*HW0, (int)(2*HW0));

    // n0 = gru08(net0, cat(mf, interp(n1)))
    k_copy16<<<CDIV((int)(128*HW0), 256), 256, 0, s>>>(mf, gx, (int)(128*HW0));
    k_interp<<<CDIV((int)(128*HW0), 256), 256, 0, s>>>(n1, gx + (size_t)128*HW0, 128, 48, 64, 96, 128);
    gru_step(s, wp, g08, n0, gx, 256, inpb[0], 96, 128, n0n, SC);
    net_cur[0] = 1 - net_cur[0]; n0 = n0n;

    // flow delta + upsampling mask
    conv(s, wp, fh1, n0, dbuf, 1, 96, 128, 1, 1.0f, 1, d1, d2);
    conv(s, wp, fh2, dbuf, delta, 1, 96, 128, 1, 1.0f, 0, d1, d2);
    conv(s, wp, mask1, n0, dbuf, 1, 96, 128, 1, 1.0f, 1, d1, d2);
    conv(s, wp, mask2, dbuf, maskb, 1, 96, 128, 1, 0.25f, 0, d1, d2);
    k_update_cx<<<CDIV((int)HW0, 256), 256, 0, s>>>(cx, delta, (int)HW0);

    if (it == ITERS - 1)
      k_upsample<<<CDIV(384*512, 256), 256, 0, s>>>(cx, maskb, (float*)d_out, 96, 128);
  }
}
